// DWSABlock_18107582120767
// MI455X (gfx1250) — matrix operand order repaired and checked by reading, not compiled, not run on hardware
//
#include <hip/hip_runtime.h>
#include <cstdint>

typedef __attribute__((ext_vector_type(16))) _Float16 v16h;
typedef __attribute__((ext_vector_type(8)))  _Float16 v8h;
typedef __attribute__((ext_vector_type(8)))  float    v8f;

#define HW 16384  // 128*128 spatial per channel plane

// ---------------- workspace layout (bytes) ----------------
#define XRED_OFF 0            // 4096 win * 64 tok * 64 ch * 2B = 33,554,432
#define ORED_OFF 33554432     // same size
#define WSW_OFF  67108864     // folded weights region
#define WIN_OFF   0           // W_in   f16 [64][256]  (BN-in folded)
#define WOUT_OFF  32768       // W_out' f16 [256][64]  (BN-out + sigmoid(alpha) folded)
#define WQ_OFF    65536       // W_q    f16 [32][64]   (SCALE folded)
#define WK_OFF    69632       // W_k    f16 [32][64]
#define WV_OFF    73728       // W_v    f16 [32][64]   (block-diag expanded)
#define WO_OFF    77824       // W_o    f16 [64][32]   (block-diag expanded)
#define BIN_OFF   81920       // b_in  f32 [64]
#define QB_OFF    82176       // qb[32] kb[32] vb[32] ob[64]  f32 contiguous
#define BO_OFF    82816       // bo    f32 [256]

// ---------------- WMMA helpers (CDNA5 f16, 16x16x32, fp32 acc) ----------------
__device__ __forceinline__ v8f wmma_f16(v16h a, v16h b, v8f c) {
  return __builtin_amdgcn_wmma_f32_16x16x32_f16(false, a, false, b, (short)0, c,
                                                false, false);
}

// A fragment 16x32 from row-major f16 [16][ld] at base.
// Lane L: row m=L%16; halves j -> K = (L>=16?16:0)+j
__device__ __forceinline__ v16h load_a_f16(const _Float16* base, int ld, int lane) {
  const int m = lane & 15, hi = lane >> 4;
  const int b1 = hi ? 8 : 0, b2 = hi ? 24 : 16;
  const _Float16* p = base + m * ld;
  v16h a;
#pragma unroll
  for (int j = 0; j < 8; ++j) { a[j] = p[hi * 16 + j]; a[8 + j] = p[hi * 16 + 8 + j]; }
  return a;
}

// B fragment 32x16 (KxN) from storage [n][k] (K contiguous per column), base at (k0,n0).
// Lane L: column n=L%16; halves t -> K = (L>=16?16:0)+t
__device__ __forceinline__ v16h load_b_kc(const _Float16* base, int ldn, int lane) {
  const int n = lane & 15, hi = lane >> 4;
  const _Float16* p = base + n * ldn + hi * 16;
  v16h b;
#pragma unroll
  for (int t = 0; t < 16; ++t) b[t] = p[t];
  return b;
}

// ---------------- Kernel A: fold all pointwise params into dense f16 weights ----------------
__global__ void prep_kernel(
    const float* __restrict__ bn_in_gamma, const float* __restrict__ bn_in_beta,
    const float* __restrict__ bn_in_mean,  const float* __restrict__ bn_in_var,
    const float* __restrict__ in_proj_w,   const float* __restrict__ qk_base_w,
    const float* __restrict__ q_head_w,    const float* __restrict__ q_head_b,
    const float* __restrict__ k_head_w,    const float* __restrict__ k_head_b,
    const float* __restrict__ v_w,         const float* __restrict__ v_b,
    const float* __restrict__ o_w,         const float* __restrict__ o_b,
    const float* __restrict__ out_proj_w,
    const float* __restrict__ bn_out_gamma, const float* __restrict__ bn_out_beta,
    const float* __restrict__ bn_out_mean,  const float* __restrict__ bn_out_var,
    const float* __restrict__ alpha, char* __restrict__ wsw)
{
  const int tid = threadIdx.x;
  _Float16* w_in  = (_Float16*)(wsw + WIN_OFF);
  _Float16* w_out = (_Float16*)(wsw + WOUT_OFF);
  _Float16* wq = (_Float16*)(wsw + WQ_OFF);
  _Float16* wk = (_Float16*)(wsw + WK_OFF);
  _Float16* wv = (_Float16*)(wsw + WV_OFF);
  _Float16* wo = (_Float16*)(wsw + WO_OFF);
  float* b_in = (float*)(wsw + BIN_OFF);
  float* qb = (float*)(wsw + QB_OFF);
  float* kb = qb + 32; float* vb = qb + 64; float* ob = qb + 96;
  float* bo = (float*)(wsw + BO_OFF);
  const float s = 1.f / (1.f + __expf(-alpha[0]));
  const float SCALE = 0.35355339059327373f;  // (MID/NHEADS)^-0.5 = 8^-0.5

  for (int i = tid; i < 64 * 256; i += 256) {          // W_in = in_proj * bn_in scale
    int c = i & 255;
    float inv = bn_in_gamma[c] * rsqrtf(bn_in_var[c] + 1e-5f);
    w_in[i] = (_Float16)(in_proj_w[i] * inv);
  }
  if (tid < 64) {                                      // b_in = in_proj @ bn_in shift
    float a = 0.f;
    for (int c = 0; c < 256; ++c) {
      float inv = bn_in_gamma[c] * rsqrtf(bn_in_var[c] + 1e-5f);
      a += in_proj_w[tid * 256 + c] * (bn_in_beta[c] - bn_in_mean[c] * inv);
    }
    b_in[tid] = a;
  }
  for (int i = tid; i < 32 * 64; i += 256) {           // Wq = SCALE*Q@Wbase ; Wk = K@Wbase ; Wv blockdiag
    int o = i >> 6, c = i & 63, g = c >> 4, ig = c & 15;
    float aq = 0.f, ak = 0.f;
    for (int j = 0; j < 8; ++j) {
      float wb = qk_base_w[g * 128 + j * 16 + ig];
      aq += q_head_w[o * 32 + g * 8 + j] * wb;
      ak += k_head_w[o * 32 + g * 8 + j] * wb;
    }
    wq[i] = (_Float16)(SCALE * aq);
    wk[i] = (_Float16)ak;
    wv[i] = ((o >> 3) == g) ? (_Float16)v_w[g * 128 + (o & 7) * 16 + ig] : (_Float16)0.f;
  }
  for (int i = tid; i < 64 * 32; i += 256) {           // Wo blockdiag expand
    int oc = i >> 5, m = i & 31, g = oc >> 4;
    wo[i] = ((m >> 3) == g) ? (_Float16)o_w[g * 128 + (oc & 15) * 8 + (m & 7)] : (_Float16)0.f;
  }
  for (int i = tid; i < 256 * 64; i += 256) {          // W_out' = s * bn_out_scale * out_proj
    int co = i >> 6;
    float inv = bn_out_gamma[co] * rsqrtf(bn_out_var[co] + 1e-5f);
    w_out[i] = (_Float16)(s * inv * out_proj_w[i]);
  }
  if (tid < 256) {
    float inv = bn_out_gamma[tid] * rsqrtf(bn_out_var[tid] + 1e-5f);
    bo[tid] = s * (bn_out_beta[tid] - bn_out_mean[tid] * inv);
  }
  if (tid < 32) { qb[tid] = SCALE * q_head_b[tid]; kb[tid] = k_head_b[tid]; vb[tid] = v_b[tid]; }
  if (tid < 64) ob[tid] = o_b[tid];
}

// ---------------- Kernel B: x_red = W_in @ bn(x), written window-major f16 ----------------
// Grid: 1024 blocks x 256 threads; each WG = 4 windows (N=256 tokens), M=64, K=256.
__global__ __launch_bounds__(256) void in_gemm_kernel(const float* __restrict__ x,
                                                      const char* __restrict__ wsw,
                                                      _Float16* __restrict__ xred)
{
  __shared__ __align__(16) _Float16 sW[64 * 256];   // 32KB folded W_in
  __shared__ __align__(16) _Float16 sX[256 * 40];   // 4win*64tok rows, K-slice of 32 ch (pad 40)
  __shared__ float sBias[64];
  const int tid = threadIdx.x;
  {
    const uint32_t* s = (const uint32_t*)(wsw + WIN_OFF);
    uint32_t* d = (uint32_t*)sW;
#pragma unroll
    for (int i = 0; i < 32; ++i) d[tid + 256 * i] = s[tid + 256 * i];
    if (tid < 64) sBias[tid] = ((const float*)(wsw + BIN_OFF))[tid];
  }
  const int w = tid >> 5, lane = tid & 31;
  const int wi = w >> 1, th = w & 1;     // wave's window (0..3) and token half (0..1)
  const int winBase = blockIdx.x * 4;
  // cooperative loader coords: one (window, token) row per thread
  const int wiL = tid >> 6, tL = tid & 63;
  const int gwinL = winBase + wiL;
  const int bL = gwinL >> 8, whL = (gwinL & 255) >> 4, wwL = gwinL & 15;
  const float* xbase = x + (size_t)bL * 256 * HW
                         + (size_t)(whL * 8 + (tL >> 3)) * 128 + (wwL * 8 + (tL & 7));
  _Float16* xrow = sX + (wiL * 64 + tL) * 40;

  v8f acc[4][2] = {};
  for (int kk = 0; kk < 8; ++kk) {
    __syncthreads();
#pragma unroll
    for (int c = 0; c < 32; ++c)
      xrow[c] = (_Float16)xbase[(size_t)(kk * 32 + c) * HW];
    __syncthreads();
    v16h bf0 = load_b_kc(sX + (wi * 64 + th * 32 + 0)  * 40, 40, lane);
    v16h bf1 = load_b_kc(sX + (wi * 64 + th * 32 + 16) * 40, 40, lane);
#pragma unroll
    for (int mt = 0; mt < 4; ++mt) {
      v16h af = load_a_f16(sW + (mt * 16) * 256 + kk * 32, 256, lane);
      acc[mt][0] = wmma_f16(af, bf0, acc[mt][0]);
      acc[mt][1] = wmma_f16(af, bf1, acc[mt][1]);
    }
  }
  const int n = lane & 15, hi = lane >> 4;
  _Float16* outw = xred + (size_t)(winBase + wi) * 4096;   // [tok 64][chan 64]
#pragma unroll
  for (int mt = 0; mt < 4; ++mt)
#pragma unroll
    for (int nt = 0; nt < 2; ++nt) {
      const int tok = th * 32 + nt * 16 + n;
      const int ch  = mt * 16 + hi * 8;
      v8h tmp;
#pragma unroll
      for (int j = 0; j < 8; ++j) tmp[j] = (_Float16)(acc[mt][nt][j] + sBias[ch + j]);
      *(v8h*)(outw + tok * 64 + ch) = tmp;
    }
}

// ---------------- Kernel C: per-window attention, all matmuls via WMMA ----------------
// Grid: 4096 blocks x 128 threads (4 waves). LDS regions are phase-overlapped (~38KB).
__global__ __launch_bounds__(128) void attn_kernel(const _Float16* __restrict__ xred,
                                                   const char* __restrict__ wsw,
                                                   _Float16* __restrict__ ored)
{
  __shared__ __align__(16) char smem[38016];
  _Float16* xw   = (_Float16*)smem;               // [tok 64][ch 64]       (phase 1)
  _Float16* Wqkv = (_Float16*)(smem + 8192);      // [3][32][64]           (phase 1)
  _Float16* Sb   = (_Float16*)smem;               // scores/probs [64][64] (phase 2+, overlays xw)
  float*    attm = (float*)(smem + 8192);         // [32][64]              (phase 3, overlays Wqkv)
  _Float16* qkv  = (_Float16*)(smem + 20480);     // q/k/v f16 [3][32][64]
  _Float16* Wo   = (_Float16*)(smem + 32768);     // [64][32]
  float*    sb   = (float*)(smem + 36864);        // qb kb vb ob (160 f32)

  const int tid = threadIdx.x;
  const int w = tid >> 5, lane = tid & 31;
  const int win = blockIdx.x;

  {  // stage window activations + folded weights
    const uint32_t* s0 = (const uint32_t*)(xred + (size_t)win * 4096);
    uint32_t* d0 = (uint32_t*)xw;
#pragma unroll
    for (int i = 0; i < 16; ++i) d0[tid + 128 * i] = s0[tid + 128 * i];
    const uint32_t* s1 = (const uint32_t*)(wsw + WQ_OFF);
    uint32_t* d1 = (uint32_t*)Wqkv;
    for (int i = tid; i < 3072; i += 128) d1[i] = s1[i];
    const uint32_t* s2 = (const uint32_t*)(wsw + WO_OFF);
    uint32_t* d2 = (uint32_t*)Wo;
    for (int i = tid; i < 1024; i += 128) d2[i] = s2[i];
    if (tid < 160) sb[tid] = ((const float*)(wsw + QB_OFF))[tid];
  }
  __syncthreads();

  // ---- phase 1: q,k,v = W{q,k,v}(32x64) @ xw^T(64x64) ; 24 tiles over 4 waves ----
  {
    const int n = lane & 15, hi = lane >> 4;
#pragma unroll
    for (int ti = 0; ti < 6; ++ti) {
      int tt = w * 6 + ti;
      int mat = tt >> 3, r = tt & 7, mt = r >> 2, nt = r & 3;
      const _Float16* Wm = Wqkv + mat * 2048;
      v8f acc = {};
#pragma unroll
      for (int kk = 0; kk < 2; ++kk) {
        v16h af = load_a_f16(Wm + (mt * 16) * 64 + kk * 32, 64, lane);
        v16h bf = load_b_kc(xw + (nt * 16) * 64 + kk * 32, 64, lane);
        acc = wmma_f16(af, bf, acc);
      }
      const float* bias = sb + mat * 32;
      _Float16* dst = qkv + mat * 2048;
#pragma unroll
      for (int j = 0; j < 8; ++j) {
        int vc = mt * 16 + hi * 8 + j;
        dst[vc * 64 + nt * 16 + n] = (_Float16)(acc[j] + bias[vc]);
      }
    }
  }
  __syncthreads();

  const _Float16* q16 = qkv;
  const _Float16* k16 = qkv + 2048;
  const _Float16* vv16 = qkv + 4096;
  const int n = lane & 15, hi = lane >> 4;

  for (int h = 0; h < 4; ++h) {
    const int t1t = w;  // each wave owns one 16-row band of tokens
    // ---- phase 2: S = q_h^T k_h  (head dim 8 zero-padded to K=32) ----
#pragma unroll
    for (int t2t = 0; t2t < 4; ++t2t) {
      v16h af, bf;
      const int b1 = hi ? 8 : 0;
#pragma unroll
      for (int j = 0; j < 8; ++j) {
        int K1 = b1 + j;
        af[j]     = (K1 < 8) ? q16[(h * 8 + K1) * 64 + t1t * 16 + (lane & 15)] : (_Float16)0.f;
        af[8 + j] = (_Float16)0.f;  // K >= 16 slots unused
      }
#pragma unroll
      for (int t2 = 0; t2 < 16; ++t2) {
        int K = hi * 16 + t2;
        bf[t2] = (K < 8) ? k16[(h * 8 + K) * 64 + t2t * 16 + n] : (_Float16)0.f;
      }
      v8f acc = {};
      acc = wmma_f16(af, bf, acc);
#pragma unroll
      for (int j = 0; j < 8; ++j)
        Sb[(t1t * 16 + hi * 8 + j) * 64 + t2t * 16 + n] = (_Float16)acc[j];
    }
    __syncthreads();
    // ---- softmax over 64 rows, in place (scores -> probs, f16) ----
    if (tid < 64) {
      _Float16* row = Sb + tid * 64;
      float mx = -1e30f;
      for (int i = 0; i < 64; ++i) mx = fmaxf(mx, (float)row[i]);
      float sum = 0.f;
      for (int i = 0; i < 64; ++i) { float e = __expf((float)row[i] - mx); sum += e; row[i] = (_Float16)e; }
      float r = 1.f / sum;
      for (int i = 0; i < 64; ++i) row[i] = (_Float16)((float)row[i] * r);
    }
    __syncthreads();
    // ---- phase 3a: att[t1][d] = P @ v_h^T  (N = d, padded 8->16) ----
    {
      v8f acc = {};
#pragma unroll
      for (int kk = 0; kk < 2; ++kk) {
        v16h af = load_a_f16(Sb + (t1t * 16) * 64 + kk * 32, 64, lane);
        v16h bf;
#pragma unroll
        for (int t2 = 0; t2 < 16; ++t2)
          bf[t2] = (n < 8) ? vv16[(h * 8 + n) * 64 + kk * 32 + hi * 16 + t2] : (_Float16)0.f;
        acc = wmma_f16(af, bf, acc);
      }
      if (n < 8) {
#pragma unroll
        for (int j = 0; j < 8; ++j)
          attm[(h * 8 + n) * 64 + t1t * 16 + hi * 8 + j] = acc[j];
      }
    }
    __syncthreads();
  }

  // ---- phase 3b: out_red = Wo(64x32) @ attm(32x64) + o_b, K=32 in one WMMA ----
  {
    const float* ob = sb + 96;
    _Float16* outw = ored + (size_t)win * 4096;
    const int mt = w;
    v16h af = load_a_f16(Wo + (mt * 16) * 32, 32, lane);
#pragma unroll
    for (int nt = 0; nt < 4; ++nt) {
      v16h bf;
#pragma unroll
      for (int t2 = 0; t2 < 16; ++t2)
        bf[t2] = (_Float16)attm[(hi * 16 + t2) * 64 + nt * 16 + n];
      v8f acc = {};
      acc = wmma_f16(af, bf, acc);
      const int ch = mt * 16 + hi * 8;
      v8h tmp;
#pragma unroll
      for (int j = 0; j < 8; ++j) tmp[j] = (_Float16)(acc[j] + ob[ch + j]);
      *(v8h*)(outw + (nt * 16 + n) * 64 + ch) = tmp;
    }
  }
}

// ---------------- Kernel D: out = x + W_out' @ out_red + b_out'  ----------------
// Grid: 2048 blocks x 256 threads; WG = 2 windows, M=256, K=64; residual fused.
__global__ __launch_bounds__(256) void out_gemm_kernel(const _Float16* __restrict__ ored,
                                                       const char* __restrict__ wsw,
                                                       const float* __restrict__ x,
                                                       float* __restrict__ out)
{
  __shared__ __align__(16) _Float16 sW[256 * 64];  // 32KB W_out'
  __shared__ float sBo[256];
  const int tid = threadIdx.x;
  {
    const uint32_t* s = (const uint32_t*)(wsw + WOUT_OFF);
    uint32_t* d = (uint32_t*)sW;
#pragma unroll
    for (int i = 0; i < 32; ++i) d[tid + 256 * i] = s[tid + 256 * i];
    sBo[tid] = ((const float*)(wsw + BO_OFF))[tid];
  }
  __syncthreads();
  const int w = tid >> 5, lane = tid & 31;
  const int wi = w >> 2, nt = w & 3;
  const int win = blockIdx.x * 2 + wi;
  const _Float16* bsrc = ored + (size_t)win * 4096;   // [tok][oc] f16, K contiguous
  v8f acc[16] = {};
#pragma unroll
  for (int kk = 0; kk < 2; ++kk) {
    v16h bf = load_b_kc(bsrc + (nt * 16) * 64 + kk * 32, 64, lane);  // direct from global
#pragma unroll
    for (int mt = 0; mt < 16; ++mt) {
      v16h af = load_a_f16(sW + (mt * 16) * 64 + kk * 32, 64, lane);
      acc[mt] = wmma_f16(af, bf, acc[mt]);
    }
  }
  const int n = lane & 15, hi = lane >> 4;
  const int b = win >> 8, wh = (win & 255) >> 4, ww = win & 15;
  const int tok = nt * 16 + n;
  const size_t sp = (size_t)(wh * 8 + (tok >> 3)) * 128 + (ww * 8 + (tok & 7));
  const float* xb = x + (size_t)b * 256 * HW + sp;
  float* ob2 = out + (size_t)b * 256 * HW + sp;
#pragma unroll
  for (int mt = 0; mt < 16; ++mt) {
    const int co = mt * 16 + hi * 8;
#pragma unroll
    for (int j = 0; j < 8; ++j) {
      const size_t off = (size_t)(co + j) * HW;
      ob2[off] = xb[off] + acc[mt][j] + sBo[co + j];
    }
  }
}

extern "C" void kernel_launch(void* const* d_in, const int* in_sizes, int n_in,
                              void* d_out, int out_size, void* d_ws, size_t ws_size,
                              hipStream_t stream) {
  (void)in_sizes; (void)n_in; (void)out_size; (void)ws_size;
  const float* x = (const float*)d_in[0];
  char* ws = (char*)d_ws;
  _Float16* xred = (_Float16*)(ws + XRED_OFF);
  _Float16* ored = (_Float16*)(ws + ORED_OFF);
  char* wsw = ws + WSW_OFF;

  prep_kernel<<<1, 256, 0, stream>>>(
      (const float*)d_in[1],  (const float*)d_in[2],  (const float*)d_in[3],
      (const float*)d_in[4],  (const float*)d_in[5],  (const float*)d_in[6],
      (const float*)d_in[7],  (const float*)d_in[8],  (const float*)d_in[9],
      (const float*)d_in[10], (const float*)d_in[11], (const float*)d_in[12],
      (const float*)d_in[13], (const float*)d_in[14], (const float*)d_in[15],
      (const float*)d_in[16], (const float*)d_in[17], (const float*)d_in[18],
      (const float*)d_in[19], (const float*)d_in[20], wsw);

  in_gemm_kernel<<<1024, 256, 0, stream>>>(x, wsw, xred);
  attn_kernel<<<4096, 128, 0, stream>>>(xred, wsw, ored);
  out_gemm_kernel<<<2048, 256, 0, stream>>>(ored, wsw, x, (float*)d_out);
}
